// CustomRNNLayer_31911607009504
// MI455X (gfx1250) — matrix operand order repaired and checked by reading, not compiled, not run on hardware
//
#include <hip/hip_runtime.h>
#include <hip/hip_bf16.h>
#include <stdint.h>

#define B_  64
#define S_  1024
#define D_  512
#define H_  512
#define K_  1024
#define NG_ 2048   // 4 gates * H

typedef __attribute__((ext_vector_type(16))) __bf16 v16bf;
typedef __attribute__((ext_vector_type(8)))  float  v8f;

struct bfrag { uint4 lo, hi; };   // 32 bytes == v16bf

__device__ __forceinline__ unsigned short f2bf(float f) {
    union { float f; unsigned int u; } v; v.f = f;
    unsigned int u = v.u;
    // round-to-nearest-even truncation to bf16
    return (unsigned short)((u + 0x7FFFu + ((u >> 16) & 1u)) >> 16);
}

__device__ __forceinline__ float sigmoidf_(float v) {
    return 1.0f / (1.0f + __expf(-v));
}

// CDNA5 async global->LDS b128 copy (ASYNCcnt-tracked, no data VGPRs).
// INST_OFFSET is added to BOTH the global and the LDS address, so one
// address pair covers a contiguous run of 16B beats.
#define ASYNC_CP16(ldsaddr, gptr, imm_off)                                   \
    asm volatile("global_load_async_to_lds_b128 %0, %1, off offset:" #imm_off \
                 :: "v"(ldsaddr), "v"(gptr) : "memory")

__device__ __forceinline__ void wait_async0() {
    asm volatile("s_wait_asynccnt 0x0" ::: "memory");
}

// ---------------------------------------------------------------------------
// One-time: convert the whole x tensor to bf16 (x never changes across steps).
// ---------------------------------------------------------------------------
__global__ __launch_bounds__(256)
void convert_x(const float* __restrict__ x, unsigned short* __restrict__ xb)
{
    size_t i = (size_t)blockIdx.x * 256 + threadIdx.x;   // float4 index
    float4 f = ((const float4*)x)[i];
    uint2 p;
    p.x = (unsigned int)f2bf(f.x) | ((unsigned int)f2bf(f.y) << 16);
    p.y = (unsigned int)f2bf(f.z) | ((unsigned int)f2bf(f.w) << 16);
    ((uint2*)xb)[i] = p;
}

// ---------------------------------------------------------------------------
// One-time: pack the four gate weight matrices into one bf16 [2048][1024]
// matrix (gate order f,i,c,o), k contiguous == WMMA B-fragment order.
// ---------------------------------------------------------------------------
__global__ __launch_bounds__(256)
void convert_weights(const float* __restrict__ Wf, const float* __restrict__ Wi,
                     const float* __restrict__ Wc, const float* __restrict__ Wo,
                     unsigned short* __restrict__ Wb)
{
    int idx = blockIdx.x * 256 + threadIdx.x;       // 0 .. 2048*1024-1
    int row = idx >> 10;
    int k   = idx & 1023;
    const float* src = (row < 512) ? Wf : (row < 1024) ? Wi : (row < 1536) ? Wc : Wo;
    Wb[idx] = f2bf(src[(size_t)(row & 511) * K_ + k]);
}

__global__ __launch_bounds__(256)
void init_state(float* __restrict__ c, unsigned short* __restrict__ hb)
{
    int idx = blockIdx.x * 256 + threadIdx.x;       // 0 .. 32767
    c[idx]  = 0.0f;
    hb[idx] = 0;                                     // bf16 +0.0
}

// ---------------------------------------------------------------------------
// Per-timestep gate GEMM, split-K along the [x | h] boundary.
//   gridDim.x = 128 : blockIdx = { kpart (bit 0), N-slab (bits 6:1) }
// Block = 256 threads = 8 waves = 4 M-tiles x 2 N-tiles of 16x16.
// Double-buffered async global->LDS staging: chunk i+1's copies are in
// flight (ASYNCcnt) while chunk i's WMMAs run; one barrier per chunk.
// ---------------------------------------------------------------------------
__global__ __launch_bounds__(256)
void lstm_gate_gemm(const unsigned short* __restrict__ xb,   // [64][1024][512] bf16
                    const unsigned short* __restrict__ hb,   // [64][512] bf16
                    const unsigned short* __restrict__ Wb,   // [2048][1024] bf16
                    float* __restrict__ gates,               // [2][64][2048] f32
                    int t)
{
    __shared__ unsigned short As[2][B_][136];   // double-buffered, 2 x 17.4 KB
    __shared__ unsigned short Bs[2][32][136];   //                  2 x  8.7 KB

    const int tid   = threadIdx.x;
    const int lane  = tid & 31;
    const int wave  = tid >> 5;
    const int mt    = wave & 3;                  // M tile 0..3
    const int nt    = wave >> 2;                 // N tile 0..1 in the slab
    const int kpart = blockIdx.x & 1;
    const int n0    = (blockIdx.x >> 1) * 32;
    const int kbase = kpart * D_;

    const int lr = lane & 15;                    // row within 16x16 tile
    const int kh = lane >> 4;                    // K-half this lane supplies

    v8f acc = {};

    // cooperative staging roles
    const int arow = tid & 63;                   // batch row 0..63
    const int acg  = tid >> 6;                   // A col group 0..3 (32 cols)
    const int brow = tid & 31;                   // W row within 32-row slab
    const int bcg  = tid >> 5;                   // B col group 0..7 (16 cols)

    // per-block uniform A base: this K-half touches only xb or only hb
    const unsigned short* abase = kpart
        ? (hb + (size_t)arow * H_ - D_)                            // index by k-D_
        : (xb + ((size_t)arow * S_ + (size_t)t) * D_);             // index by k
    const unsigned short* bbase = Wb + (size_t)(n0 + brow) * K_ + bcg * 16;

    // this thread's LDS deposit addresses in each buffer
    const unsigned lds_a0 = (unsigned)(uintptr_t)&As[0][arow][acg * 32];
    const unsigned lds_a1 = (unsigned)(uintptr_t)&As[1][arow][acg * 32];
    const unsigned lds_b0 = (unsigned)(uintptr_t)&Bs[0][brow][bcg * 16];
    const unsigned lds_b1 = (unsigned)(uintptr_t)&Bs[1][brow][bcg * 16];

    // issue one chunk's async copies: 4x16B of A + 2x16B of B per thread
    auto issue_chunk = [&](int kc, int bufi) {
        const unsigned short* ga = abase + kc + acg * 32;
        const unsigned short* gb = bbase + kc;
        const unsigned la = bufi ? lds_a1 : lds_a0;
        const unsigned lb = bufi ? lds_b1 : lds_b0;
        ASYNC_CP16(la, ga, 0);
        ASYNC_CP16(la, ga, 16);
        ASYNC_CP16(la, ga, 32);
        ASYNC_CP16(la, ga, 48);
        ASYNC_CP16(lb, gb, 0);
        ASYNC_CP16(lb, gb, 16);
    };

    issue_chunk(kbase, 0);                       // prologue: chunk 0 in flight

    #pragma unroll
    for (int i = 0; i < 4; ++i) {                // 4 chunks of K=128
        const int bufi = i & 1;
        wait_async0();                           // my chunk-i deposits landed
        __syncthreads();                         // all deposits visible; buf re-fill safe
        if (i + 1 < 4)
            issue_chunk(kbase + (i + 1) * 128, bufi ^ 1);

        const unsigned short* arow_p = &As[bufi][mt * 16 + lr][0];
        const unsigned short* brow_p = &Bs[bufi][nt * 16 + lr][0];
        #pragma unroll
        for (int ks = 0; ks < 128; ks += 32) {
            // A fragment: lane lr = row M; elems 0..7 = K[ks+kh*8..],
            //             elems 8..15 = K[ks+16+kh*8..]   (ISA 16-bit A layout)
            bfrag fa, fb;
            fa.lo = *(const uint4*)(arow_p + ks + kh * 8);
            fa.hi = *(const uint4*)(arow_p + ks + 16 + kh * 8);
            // B fragment: lane lr = col N; elems 0..7 = K[ks+kh*8..], elems 8..15 = K[ks+16+kh*8..]
            fb.lo = *(const uint4*)(brow_p + ks + kh * 8);
            fb.hi = *(const uint4*)(brow_p + ks + 16 + kh * 8);
            v16bf a = __builtin_bit_cast(v16bf, fa);
            v16bf b = __builtin_bit_cast(v16bf, fb);
            acc = __builtin_amdgcn_wmma_f32_16x16x32_bf16(
                      /*neg_a=*/false, a, /*neg_b=*/false, b,
                      /*c_mod=*/(short)0, acc, /*reuse_a=*/false, /*reuse_b=*/false);
        }
    }

    // store 16x16 f32 tile: VGPR v holds M=v (lanes 0-15) / v+8 (lanes 16-31)
    float* gpart = gates + (size_t)kpart * (B_ * NG_);
    const int ng = n0 + nt * 16 + lr;
    #pragma unroll
    for (int v = 0; v < 8; ++v) {
        const int m = mt * 16 + v + 8 * kh;
        gpart[(size_t)m * NG_ + ng] = acc[v];
    }
}

// ---------------------------------------------------------------------------
// Fused: sum split-K partials + bias, sigmoid/tanh, c/h update,
// h -> output (f32) and h -> hb (bf16) for the next step's GEMM.
// ---------------------------------------------------------------------------
__global__ __launch_bounds__(256)
void lstm_pointwise(const float* __restrict__ gates,        // [2][64][2048]
                    const float* __restrict__ bfv, const float* __restrict__ biv,
                    const float* __restrict__ bcv, const float* __restrict__ bov,
                    float* __restrict__ cbuf,
                    unsigned short* __restrict__ hb,
                    float* __restrict__ out, int t)
{
    int idx = blockIdx.x * 256 + threadIdx.x;   // 0 .. 32767
    int b = idx >> 9;
    int n = idx & 511;
    const float* g0 = gates + (size_t)b * NG_;
    const float* g1 = g0 + (size_t)(B_ * NG_);
    float fg = sigmoidf_(g0[n]        + g1[n]        + bfv[n]);
    float ig = sigmoidf_(g0[512 + n]  + g1[512 + n]  + biv[n]);
    float cg = tanhf    (g0[1024 + n] + g1[1024 + n] + bcv[n]);
    float og = sigmoidf_(g0[1536 + n] + g1[1536 + n] + bov[n]);
    float c  = fg * cbuf[idx] + ig * cg;
    float h  = og * tanhf(c);
    cbuf[idx] = c;
    hb[idx]   = f2bf(h);
    out[(size_t)b * (size_t)(S_ * H_) + (size_t)t * H_ + n] = h;
}

// h_last == outputs[:, S-1, :], already written; just gather it.
__global__ __launch_bounds__(256)
void copy_hlast(const float* __restrict__ out, float* __restrict__ dst)
{
    int idx = blockIdx.x * 256 + threadIdx.x;   // 0 .. 32767
    int b = idx >> 9;
    int n = idx & 511;
    dst[idx] = out[(size_t)b * (size_t)(S_ * H_) + (size_t)(S_ - 1) * H_ + n];
}

// ---------------------------------------------------------------------------
extern "C" void kernel_launch(void* const* d_in, const int* in_sizes, int n_in,
                              void* d_out, int out_size, void* d_ws, size_t ws_size,
                              hipStream_t stream)
{
    const float* x   = (const float*)d_in[0];
    const float* Wf  = (const float*)d_in[1];
    const float* bfv = (const float*)d_in[2];
    const float* Wi  = (const float*)d_in[3];
    const float* biv = (const float*)d_in[4];
    const float* Wc  = (const float*)d_in[5];
    const float* bcv = (const float*)d_in[6];
    const float* Wo  = (const float*)d_in[7];
    const float* bov = (const float*)d_in[8];
    float* out = (float*)d_out;

    // workspace carve-up (~70 MB)
    char* ws = (char*)d_ws;
    unsigned short* xb = (unsigned short*)ws;                     // 64 MiB bf16 x
    char* p = ws + (size_t)B_ * S_ * D_ * sizeof(unsigned short);
    unsigned short* Wb = (unsigned short*)p;                      // 4 MiB bf16 weights
    p += (size_t)NG_ * K_ * sizeof(unsigned short);
    float* gates = (float*)p;                                     // 2 x 512 KiB partials
    p += (size_t)2 * B_ * NG_ * sizeof(float);
    float* cbuf = (float*)p;                                      // 128 KiB
    p += (size_t)B_ * H_ * sizeof(float);
    unsigned short* hb = (unsigned short*)p;                      // 64 KiB bf16 h

    convert_x<<<(B_ * S_ * D_ / 4) / 256, 256, 0, stream>>>(x, xb);
    convert_weights<<<(NG_ * K_) / 256, 256, 0, stream>>>(Wf, Wi, Wc, Wo, Wb);
    init_state<<<(B_ * H_) / 256, 256, 0, stream>>>(cbuf, hb);

    for (int t = 0; t < S_; ++t) {
        lstm_gate_gemm<<<128, 256, 0, stream>>>(xb, hb, Wb, gates, t);
        lstm_pointwise<<<(B_ * H_) / 256, 256, 0, stream>>>(gates, bfv, biv, bcv, bov,
                                                            cbuf, hb, out, t);
    }
    copy_hlast<<<(B_ * H_) / 256, 256, 0, stream>>>(out, out + (size_t)B_ * S_ * H_);
}
